// ReBertAttention_38397007626392
// MI455X (gfx1250) — matrix operand order repaired and checked by reading, not compiled, not run on hardware
//
#include <hip/hip_runtime.h>
#include <hip/hip_bf16.h>

#define S_LEN 4096
#define DM    1024
#define DH    64

typedef __attribute__((ext_vector_type(16))) __bf16 v16bf;
typedef __attribute__((ext_vector_type(8)))  __bf16 v8bf;
typedef __attribute__((ext_vector_type(8)))  float  v8f;

union FragB16 {
    v16bf v;
    unsigned short h[16];
    uint4 u4[2];
};

union Pack8 {
    v8bf v;
    uint4 u;
    unsigned short h[8];
};

// scalar f32 -> bf16 (native convert)
__device__ __forceinline__ unsigned short f2bf(float f) {
    __bf16 b = (__bf16)f;
    return __builtin_bit_cast(unsigned short, b);
}

// 8 contiguous f32 -> 8 packed bf16 (one uint4) via native packed converts
__device__ __forceinline__ uint4 cvt8(const float* __restrict__ p) {
    v8f f;
#pragma unroll
    for (int i = 0; i < 8; ++i) f[i] = p[i];
    Pack8 pk;
    pk.v = __builtin_convertvector(f, v8bf);
    return pk.u;
}

__device__ __forceinline__ v8f wmma_bf16(const FragB16& a, const FragB16& b, v8f c) {
    return __builtin_amdgcn_wmma_f32_16x16x32_bf16(false, a.v, false, b.v,
                                                   (short)0, c, false, false);
}

// A-fragment (16x32 bf16) from an fp32 row: elems 0..7 at K=16*kh, 8..15 at K=16*kh+8
__device__ __forceinline__ void load_a_f32(FragB16& a, const float* p0, const float* p1) {
    a.u4[0] = cvt8(p0);
    a.u4[1] = cvt8(p1);
}

// B-fragment: 16 contiguous fp32 (K = 16*kh .. 16*kh+15 of this lane's output column)
__device__ __forceinline__ void load_b_f32(FragB16& b, const float* p) {
    b.u4[0] = cvt8(p);
    b.u4[1] = cvt8(p + 8);
}

// A-fragment from a bf16 row-major row (16 contiguous elems at K = 16*kh)
__device__ __forceinline__ void load_a_bf16(FragB16& a, const unsigned short* row, int kh) {
    a.u4[0] = *(const uint4*)(row + 16 * kh);
    a.u4[1] = *(const uint4*)(row + 16 * kh + 8);
}

// B-fragment from bf16: 16 contiguous elems (32B) at K = 16*kh
__device__ __forceinline__ void load_b_bf16(FragB16& b, const unsigned short* p) {
    const uint4* q = (const uint4*)p;
    b.u4[0] = q[0];
    b.u4[1] = q[1];
}

// ---- gfx1250 async global->LDS copy (16B per lane), tracked by ASYNCcnt -----
// LDS byte offset = low 32 bits of the generic shared-aperture address.
__device__ __forceinline__ void async_b128(unsigned lds_off, const void* gsrc) {
    unsigned long long ga = (unsigned long long)(size_t)gsrc;
    asm volatile("global_load_async_to_lds_b128 %0, %1, off"
                 :: "v"(lds_off), "v"(ga) : "memory");
}

__device__ __forceinline__ unsigned lds_off32(const void* p) {
    return (unsigned)(size_t)p;
}

// ---------------- Q/K projection fused with RoPE -----------------------------
// One wave computes a full 16-row x 64-col head tile, applies RoPE in-register.
__global__ void proj_rope_kernel(const float* __restrict__ seq,   // [S][1024]
                                 const float* __restrict__ W,     // [H*64][1024]
                                 unsigned short* __restrict__ dst)// [H][S][64]
{
    const int lane = threadIdx.x & 31;
    const int wid  = threadIdx.x >> 5;
    const int h    = blockIdx.x;
    const int q0   = (blockIdx.y * 4 + wid) * 16;
    const int kh   = lane >> 4;
    const int l15  = lane & 15;

    v8f acc[4] = {};
    const float* abase = seq + (size_t)(q0 + l15) * DM;

    for (int kc = 0; kc < DM / 32; ++kc) {
        FragB16 a;
        load_a_f32(a, abase + kc * 32 + 16 * kh, abase + kc * 32 + 16 * kh + 8);
#pragma unroll
        for (int t = 0; t < 4; ++t) {
            FragB16 b;
            const int o = h * 64 + t * 16 + l15;
            load_b_f32(b, W + (size_t)o * DM + kc * 32 + 16 * kh);
            acc[t] = wmma_bf16(a, b, acc[t]);
        }
    }

    const float LN1E4 = 9.210340371976184f; // ln(10000)
#pragma unroll
    for (int r = 0; r < 8; ++r) {
        const int row = q0 + r + 8 * kh;
#pragma unroll
        for (int t = 0; t < 2; ++t) {
            const int c = t * 16 + l15;                 // 0..31
            const float invf = __expf(-LN1E4 * (float)c * (1.0f / 32.0f));
            float sn, cs;
            __sincosf((float)row * invf, &sn, &cs);
            const float x = acc[t][r], y = acc[t + 2][r];
            unsigned short* drow = dst + ((size_t)h * S_LEN + row) * 64;
            drow[c]      = f2bf(x * cs - y * sn);
            drow[c + 32] = f2bf(y * cs + x * sn);
        }
    }
}

// ---------------- V projection, stored transposed [kvh][64][S] ---------------
__global__ void proj_v_kernel(const float* __restrict__ seq,
                              const float* __restrict__ W,       // [256][1024]
                              unsigned short* __restrict__ Vt)   // [4][64][S]
{
    const int lane = threadIdx.x & 31;
    const int wid  = threadIdx.x >> 5;
    const int h    = blockIdx.x;                 // kv head
    const int q0   = (blockIdx.y * 4 + wid) * 16;
    const int kh   = lane >> 4;
    const int l15  = lane & 15;

    v8f acc[4] = {};
    const float* abase = seq + (size_t)(q0 + l15) * DM;

    for (int kc = 0; kc < DM / 32; ++kc) {
        FragB16 a;
        load_a_f32(a, abase + kc * 32 + 16 * kh, abase + kc * 32 + 16 * kh + 8);
#pragma unroll
        for (int t = 0; t < 4; ++t) {
            FragB16 b;
            const int o = h * 64 + t * 16 + l15;
            load_b_f32(b, W + (size_t)o * DM + kc * 32 + 16 * kh);
            acc[t] = wmma_bf16(a, b, acc[t]);
        }
    }
    // lane's 8 rows are consecutive positions in Vt[d][*] -> one packed 16B store
#pragma unroll
    for (int t = 0; t < 4; ++t) {
        const int d = t * 16 + l15;
        Pack8 pk;
        pk.v = __builtin_convertvector(acc[t], v8bf);
        *(uint4*)(Vt + ((size_t)h * 64 + d) * S_LEN + q0 + 8 * kh) = pk.u;
    }
}

// ---------------- Flash attention (online softmax, per-head) -----------------
// 4 waves/block share one head; K/V tiles staged once per block into LDS via
// async DMA (double-buffered), B-fragments then read with ds_load_b128.
__global__ void flash_kernel(const unsigned short* __restrict__ Qb, // [16][S][64]
                             const unsigned short* __restrict__ Kb, // [4][S][64]
                             const unsigned short* __restrict__ Vt, // [4][64][S]
                             const float* __restrict__ mask,        // [S][S]
                             unsigned short* __restrict__ attnout)  // [S][1024]
{
    __shared__ __align__(16) unsigned short Ktile[2][32 * 64]; // 2 x 4KB
    __shared__ __align__(16) unsigned short Vtile[2][64 * 32]; // 2 x 4KB
    __shared__ __align__(16) unsigned short Pbuf[4][16 * 32];  // 4KB

    const int tid  = threadIdx.x;
    const int lane = tid & 31;
    const int wid  = tid >> 5;
    const int h    = blockIdx.y;
    const int kvh  = h >> 2;
    const int q0   = (blockIdx.x * 4 + wid) * 16;
    const int kh   = lane >> 4;
    const int l15  = lane & 15;
    unsigned short* P = Pbuf[wid];

    const unsigned short* Kh = Kb + (size_t)kvh * S_LEN * 64;
    const unsigned short* Vh = Vt + (size_t)kvh * 64 * S_LEN;

    // per-thread staging slices (128 threads x 16 elems = one 32x64 tile)
    const int krow = tid >> 2, kcol = (tid & 3) * 16;       // K: 32 rows x 64
    const int vd   = tid >> 1, vj   = (tid & 1) * 16;       // V: 64 d  x 32 j

    FragB16 qa[2];
    {
        const unsigned short* qrow = Qb + ((size_t)h * S_LEN + q0 + l15) * 64;
        load_a_bf16(qa[0], qrow, kh);
        load_a_bf16(qa[1], qrow + 32, kh);
    }

    v8f acc[4] = {};
    float m[8], lsum[8];
#pragma unroll
    for (int r = 0; r < 8; ++r) { m[r] = -3.0e38f; lsum[r] = 0.0f; }

    // stage tile 0
    {
        const unsigned short* ks = Kh + (size_t)krow * 64 + kcol;
        unsigned kd = lds_off32(Ktile[0] + krow * 64 + kcol);
        async_b128(kd, ks);
        async_b128(kd + 16, ks + 8);
        const unsigned short* vs = Vh + (size_t)vd * S_LEN + vj;
        unsigned vdd = lds_off32(Vtile[0] + vd * 32 + vj);
        async_b128(vdd, vs);
        async_b128(vdd + 16, vs + 8);
    }

    for (int jt = 0; jt < S_LEN / 32; ++jt) {
        const int b = jt & 1;
        asm volatile("s_wait_asynccnt 0x0" ::: "memory");
        __syncthreads();

        // kick off DMA for the next tile while computing this one
        if (jt + 1 < S_LEN / 32) {
            const unsigned short* ks = Kh + (size_t)((jt + 1) * 32 + krow) * 64 + kcol;
            unsigned kd = lds_off32(Ktile[b ^ 1] + krow * 64 + kcol);
            async_b128(kd, ks);
            async_b128(kd + 16, ks + 8);
            const unsigned short* vs = Vh + (size_t)vd * S_LEN + (jt + 1) * 32 + vj;
            unsigned vdd = lds_off32(Vtile[b ^ 1] + vd * 32 + vj);
            async_b128(vdd, vs);
            async_b128(vdd + 16, vs + 8);
        }

        const unsigned short* Kt = Ktile[b];
        const unsigned short* Vl = Vtile[b];

        v8f s[2] = {};
        // scores: QK^T for 2 subtiles of 16 kv cols, K-dim 64 in 2 chunks
#pragma unroll
        for (int t = 0; t < 2; ++t) {
            const unsigned short* krowp = Kt + (t * 16 + l15) * 64;
#pragma unroll
            for (int c = 0; c < 2; ++c) {
                FragB16 bb;
                load_b_bf16(bb, krowp + c * 32 + 16 * kh);
                s[t] = wmma_bf16(qa[c], bb, s[t]);
            }
        }
        // scale + additive mask
#pragma unroll
        for (int t = 0; t < 2; ++t) {
            const int j = jt * 32 + t * 16 + l15;
#pragma unroll
            for (int r = 0; r < 8; ++r) {
                const int row = q0 + r + 8 * kh;
                s[t][r] = s[t][r] * 0.125f + mask[(size_t)row * S_LEN + j];
            }
        }
        // online softmax: row max/sum via butterfly over the 16-lane half
        float cand[8], p0v[8], p1v[8], rs[8], scl[8];
#pragma unroll
        for (int r = 0; r < 8; ++r) cand[r] = fmaxf(s[0][r], s[1][r]);
        for (int mk = 1; mk < 16; mk <<= 1)
#pragma unroll
            for (int r = 0; r < 8; ++r)
                cand[r] = fmaxf(cand[r], __shfl_xor(cand[r], mk, 32));
#pragma unroll
        for (int r = 0; r < 8; ++r) {
            const float nm = fmaxf(m[r], cand[r]);
            p0v[r] = __expf(s[0][r] - nm);
            p1v[r] = __expf(s[1][r] - nm);
            rs[r]  = p0v[r] + p1v[r];
            scl[r] = __expf(m[r] - nm);
            m[r] = nm;
        }
        for (int mk = 1; mk < 16; mk <<= 1)
#pragma unroll
            for (int r = 0; r < 8; ++r) rs[r] += __shfl_xor(rs[r], mk, 32);
#pragma unroll
        for (int r = 0; r < 8; ++r) lsum[r] = lsum[r] * scl[r] + rs[r];
#pragma unroll
        for (int t = 0; t < 4; ++t)
#pragma unroll
            for (int r = 0; r < 8; ++r) acc[t][r] *= scl[r];

        // transpose P (C-layout -> A-layout) through per-wave LDS
#pragma unroll
        for (int r = 0; r < 8; ++r) {
            const int row = r + 8 * kh;
            P[row * 32 + l15]      = f2bf(p0v[r]);
            P[row * 32 + 16 + l15] = f2bf(p1v[r]);
        }
        asm volatile("s_wait_dscnt 0x0" ::: "memory");
        FragB16 pa;
        load_a_bf16(pa, P + l15 * 32, kh);

        // out += P @ V  (B from LDS-staged transposed V -> contiguous)
#pragma unroll
        for (int t = 0; t < 4; ++t) {
            FragB16 vb;
            load_b_bf16(vb, Vl + (t * 16 + l15) * 32 + 16 * kh);
            acc[t] = wmma_bf16(pa, vb, acc[t]);
        }
    }
    // normalize + store head-interleaved [S][1024]
#pragma unroll
    for (int t = 0; t < 4; ++t)
#pragma unroll
        for (int r = 0; r < 8; ++r) {
            const int row = q0 + r + 8 * kh;
            attnout[(size_t)row * DM + h * 64 + t * 16 + l15] =
                f2bf(acc[t][r] / lsum[r]);
        }
}

// ---------------- Output projection ------------------------------------------
__global__ void outproj_kernel(const unsigned short* __restrict__ A, // [S][1024] bf16
                               const float* __restrict__ Wo,         // [1024][1024]
                               float* __restrict__ out)              // [S][1024]
{
    const int lane = threadIdx.x & 31;
    const int wid  = threadIdx.x >> 5;
    const int n0   = blockIdx.x * 16;
    const int q0   = (blockIdx.y * 4 + wid) * 16;
    const int kh   = lane >> 4;
    const int l15  = lane & 15;

    v8f acc = {};
    const unsigned short* arow = A + (size_t)(q0 + l15) * DM;
    const int o = n0 + l15;

    for (int kc = 0; kc < DM / 32; ++kc) {
        FragB16 a, b;
        load_a_bf16(a, arow + kc * 32, kh);
        load_b_f32(b, Wo + (size_t)o * DM + kc * 32 + 16 * kh);
        acc = wmma_bf16(a, b, acc);
    }
#pragma unroll
    for (int r = 0; r < 8; ++r)
        out[(size_t)(q0 + r + 8 * kh) * DM + n0 + l15] = acc[r];
}

extern "C" void kernel_launch(void* const* d_in, const int* in_sizes, int n_in,
                              void* d_out, int out_size, void* d_ws, size_t ws_size,
                              hipStream_t stream) {
    const float* seq  = (const float*)d_in[0];
    const float* mask = (const float*)d_in[1];
    // d_in[2] = position_ids (arange; positions computed implicitly)
    const float* wq   = (const float*)d_in[3];
    const float* wk   = (const float*)d_in[4];
    const float* wv   = (const float*)d_in[5];
    const float* wo   = (const float*)d_in[6];
    float* out        = (float*)d_out;

    char* ws = (char*)d_ws;
    unsigned short* Qb      = (unsigned short*)(ws);                    // 8 MB
    unsigned short* Kb      = (unsigned short*)(ws + (8u << 20));       // 2 MB
    unsigned short* Vtw     = (unsigned short*)(ws + (10u << 20));      // 2 MB
    unsigned short* attnout = (unsigned short*)(ws + (12u << 20));      // 8 MB

    dim3 blk(128, 1, 1);
    proj_rope_kernel<<<dim3(16, S_LEN / 64), blk, 0, stream>>>(seq, wq, Qb);
    proj_rope_kernel<<<dim3(4,  S_LEN / 64), blk, 0, stream>>>(seq, wk, Kb);
    proj_v_kernel   <<<dim3(4,  S_LEN / 64), blk, 0, stream>>>(seq, wv, Vtw);
    flash_kernel    <<<dim3(S_LEN / 64, 16), blk, 0, stream>>>(Qb, Kb, Vtw, mask, attnout);
    outproj_kernel  <<<dim3(DM / 16, S_LEN / 64), blk, 0, stream>>>(attnout, wo, out);
}
